// SimpleGRU1_3547642986742
// MI455X (gfx1250) — matrix operand order repaired and checked by reading, not compiled, not run on hardware
//
#include <hip/hip_runtime.h>
#include <hip/hip_bf16.h>
#include <math.h>

// Problem constants (from reference)
#define B_TOT 256
#define T_LEN 2048
#define D_IN  64
#define H     100
#define HPAD  128
#define NPAD  384      // 3 gates * HPAD
#define TPB   256      // 8 waves of 32
#define NWAVE 8
#define TPW   3        // N-tiles per wave (8*3 = 24 tiles = 384 cols)
#define MROWS 16       // batch rows per workgroup (one WMMA M tile)
#define XROW  68       // padded x row stride in f32 (TDM pads 4 dwords / 64)

typedef __bf16 bf16_t;
typedef __attribute__((ext_vector_type(16))) __bf16        v16bf;
typedef __attribute__((ext_vector_type(8)))  float         v8f;
typedef __attribute__((ext_vector_type(4)))  unsigned int  v4u;
typedef __attribute__((ext_vector_type(4)))  int           v4i;
typedef __attribute__((ext_vector_type(8)))  int           v8i;
typedef __attribute__((ext_vector_type(4)))  unsigned int  u32x4;

union ABFrag { v16bf v; u32x4 q[2]; };

// ---------------------------------------------------------------------------
// Tensor Data Mover: async DMA one [16 x 64] f32 x-tile (rows = batch, row
// stride = T*D elements) from global into LDS, padding the LDS destination
// by 4 dwords every 64 dwords (row stride 68 f32 -> bank-conflict-free reads).
// Tracked on TENSORcnt.
// ---------------------------------------------------------------------------
__device__ __forceinline__ void tdm_load_x_tile(const float* xg, int b0, int t,
                                                float* lds_dst) {
  unsigned lds_addr = (unsigned)(unsigned long long)lds_dst;   // low 32 bits = LDS byte offset
  unsigned long long ga = (unsigned long long)xg +
      (((unsigned long long)b0 * (unsigned long long)T_LEN +
        (unsigned long long)t) * (unsigned long long)D_IN) * 4ull;

  v4u g0;
  g0[0] = 1u;                                              // count=1, user descriptor
  g0[1] = lds_addr;                                        // lds_addr[31:0]
  g0[2] = (unsigned)ga;                                    // global_addr[31:0]
  g0[3] = (unsigned)((ga >> 32) & 0x01FFFFFFu)             // global_addr[56:32]
        | 0x80000000u;                                     // type=2 ("image") in bits 127:126

  v8i g1;
  g1[0] = (int)((2u << 16)       // data_size=2 (4 bytes), workgroup_mask=0
              | (1u << 20)       // pad_enable
              | (5u << 22)       // pad_interval: every 64 dwords
              | (3u << 25));     // pad_amount: 4 dwords
  g1[1] = (int)(D_IN << 16);     // tensor_dim0[15:0] (bits 79:48)
  g1[2] = (int)(MROWS << 16);    // tensor_dim0 hi = 0 | tensor_dim1[15:0]
  g1[3] = (int)(D_IN << 16);     // tensor_dim1 hi = 0 | tile_dim0 = 64
  g1[4] = MROWS;                 // tile_dim1 = 16, tile_dim2 = 0 (2D)
  g1[5] = T_LEN * D_IN;          // tensor_dim0_stride[31:0] (elements)
  g1[6] = 0;                     // stride0 hi | stride1 lo
  g1[7] = 0;                     // stride1 hi

  v4i g2 = {0, 0, 0, 0};
  v4i g3 = {0, 0, 0, 0};
#if defined(__clang_major__) && (__clang_major__ >= 23)
  v8i g4 = {0, 0, 0, 0, 0, 0, 0, 0};
  __builtin_amdgcn_tensor_load_to_lds(g0, g1, g2, g3, g4, 0);
#else
  __builtin_amdgcn_tensor_load_to_lds(g0, g1, g2, g3, 0);
#endif
}

__device__ __forceinline__ float fast_sigmoid(float v) {
  return 1.f / (1.f + __expf(-v));
}
__device__ __forceinline__ float fast_tanh(float v) {
  float e2 = __expf(2.f * v);
  return (e2 - 1.f) / (e2 + 1.f);
}

__global__ void __launch_bounds__(TPB, 1)
gru_fused(const float* __restrict__ x,   const float* __restrict__ W_ih,
          const float* __restrict__ b_ih, const float* __restrict__ W_hh,
          const float* __restrict__ b_hh, const float* __restrict__ W_fc,
          const float* __restrict__ b_fc, float* __restrict__ out) {
  // LDS (~58 KB). hbf stride 136 (68 dwords) and xbuf stride 68 dwords keep
  // per-lane row bases in distinct banks for the b128 fragment loads.
  __shared__ float  xbuf[2][MROWS][XROW];   // TDM destination (padded), dbl buf
  __shared__ bf16_t hbf[MROWS][136];        // bf16 h tile (A matrix, K pad 128)
  __shared__ float  hfs[MROWS][HPAD];       // fp32 hidden state
  __shared__ float  gsum[MROWS][NPAD];      // r,z: gi+gh ; n: gi
  __shared__ float  ghn[MROWS][HPAD];       // n: gh (kept separate for r*gh_n)
  __shared__ float  bih_s[NPAD];
  __shared__ float  bhh_s[NPAD];
  __shared__ float  lg[MROWS][10];

  const int tid  = threadIdx.x;
  const int wave = tid >> 5;
  const int lane = tid & 31;
  const int b0   = blockIdx.x * MROWS;

  // ---- init: h = 0, biases into padded-gate layout --------------------------
  for (int i = tid; i < MROWS * HPAD; i += TPB) hfs[i >> 7][i & 127] = 0.f;
  for (int i = tid; i < MROWS * 136; i += TPB) hbf[i / 136][i % 136] = (bf16_t)0.f;
  for (int i = tid; i < NPAD; i += TPB) {
    int g = i >> 7, j = i & 127;
    float vi = 0.f, vh = 0.f;
    if (j < H) { vi = b_ih[g * H + j]; vh = b_hh[g * H + j]; }
    bih_s[i] = vi; bhh_s[i] = vh;
  }

  // ---- weights -> register-resident bf16 B fragments ------------------------
  // B (KxN) layout, 16-bit, wave32: lane L holds column N = L&15,
  // K = { kb..kb+7 , 16+kb..16+kb+7 }, kb=(L>>4)*8, within each 32-deep K step.
  const int nloc  = lane & 15;
  const int kbase = (lane >> 4) * 8;
  v16bf bh[TPW][4];  // hidden side: K = 128 (H=100 zero-padded)
  v16bf bi[TPW][2];  // input  side: K = 64
  #pragma unroll
  for (int tt = 0; tt < TPW; ++tt) {
    const int tile = wave * TPW + tt;
    const int c = tile * 16 + nloc;       // padded gate column
    const int g = c >> 7;                 // gate index (0=r,1=z,2=n)
    const int j = c & 127;                // column within gate
    const bool vn = (j < H);
    const int row = g * H + j;            // row of W_ih / W_hh
    #pragma unroll
    for (int ks = 0; ks < 4; ++ks) {
      v16bf f;
      #pragma unroll
      for (int i = 0; i < 16; ++i) {
        int k = ks * 32 + kbase + i + (i >= 8 ? 8 : 0);
        float wv = (vn && k < H) ? W_hh[row * H + k] : 0.f;
        f[i] = (bf16_t)wv;
      }
      bh[tt][ks] = f;
    }
    #pragma unroll
    for (int ks = 0; ks < 2; ++ks) {
      v16bf f;
      #pragma unroll
      for (int i = 0; i < 16; ++i) {
        int k = ks * 32 + kbase + i + (i >= 8 ? 8 : 0);      // always < 64
        float wv = vn ? W_ih[row * D_IN + k] : 0.f;
        f[i] = (bf16_t)wv;
      }
      bi[tt][ks] = f;
    }
  }

  // ---- preload x(t=0) via TDM ----------------------------------------------
  if (wave == 0) {
    tdm_load_x_tile(x, b0, 0, &xbuf[0][0][0]);
    __builtin_amdgcn_s_wait_tensorcnt(0);
  }
  __syncthreads();

  // ---- sequential scan over time (2 barriers per step) ----------------------
  for (int t = 0; t < T_LEN; ++t) {
    const int buf = t & 1;
    if (wave == 0 && (t + 1) < T_LEN)
      tdm_load_x_tile(x, b0, t + 1, &xbuf[1 - buf][0][0]);  // async prefetch

    // A fragments (16-bit A 16x32 layout): lane L holds row M = L&15,
    // K = ks*32 + { khalf..khalf+7 , 16+khalf..16+khalf+7 }, khalf=(L>>4)*8.
    const int m = lane & 15;
    const int khalf = (lane >> 4) * 8;
    ABFrag ah[4], ax[2];
    #pragma unroll
    for (int ks = 0; ks < 4; ++ks) {
      ah[ks].q[0] = *(const u32x4*)&hbf[m][ks * 32 + khalf];
      ah[ks].q[1] = *(const u32x4*)&hbf[m][ks * 32 + 16 + khalf];
    }
    // x A-fragments: load f32 from the padded TDM tile, convert in-register.
    #pragma unroll
    for (int ks = 0; ks < 2; ++ks) {
      const float* s0 = &xbuf[buf][m][ks * 32 + khalf];
      const float* s1 = &xbuf[buf][m][ks * 32 + 16 + khalf];
      #pragma unroll
      for (int i = 0; i < 8; ++i) ax[ks].v[i]     = (bf16_t)s0[i];
      #pragma unroll
      for (int i = 0; i < 8; ++i) ax[ks].v[8 + i] = (bf16_t)s1[i];
    }

    #pragma unroll
    for (int tt = 0; tt < TPW; ++tt) {
      const int tile = wave * TPW + tt;
      const int c = tile * 16 + nloc;
      // gate index is uniform per wave for a given tt: make it scalar so the
      // store selection is an s_cbranch, not an EXEC-mask double-execution.
      const int gate_u = __builtin_amdgcn_readfirstlane(tile) >> 3;
      v8f acc_h = {0.f, 0.f, 0.f, 0.f, 0.f, 0.f, 0.f, 0.f};
      v8f acc_i = {0.f, 0.f, 0.f, 0.f, 0.f, 0.f, 0.f, 0.f};
      #pragma unroll
      for (int ks = 0; ks < 4; ++ks)
        acc_h = __builtin_amdgcn_wmma_f32_16x16x32_bf16(
            false, ah[ks].v, false, bh[tt][ks], (short)0, acc_h, false, false);
      #pragma unroll
      for (int ks = 0; ks < 2; ++ks)
        acc_i = __builtin_amdgcn_wmma_f32_16x16x32_bf16(
            false, ax[ks].v, false, bi[tt][ks], (short)0, acc_i, false, false);

      // C/D layout: lanes 0-15 -> M=v, lanes 16-31 -> M=8+v; N = c
      const int mrow0 = (lane >> 4) * 8;
      if (gate_u < 2) {
        #pragma unroll
        for (int vv = 0; vv < 8; ++vv)
          gsum[mrow0 + vv][c] = acc_i[vv] + acc_h[vv];
      } else {
        #pragma unroll
        for (int vv = 0; vv < 8; ++vv) {
          gsum[mrow0 + vv][c] = acc_i[vv];
          ghn[mrow0 + vv][c & 127] = acc_h[vv];
        }
      }
    }
    __syncthreads();

    // elementwise GRU cell (16*128 padded slots / 256 threads = 8 iters)
    #pragma unroll
    for (int it = 0; it < (MROWS * HPAD) / TPB; ++it) {
      int idx = tid + it * TPB;
      int mm = idx >> 7, j = idx & 127;
      if (j < H) {
        float rg = gsum[mm][j]       + bih_s[j]       + bhh_s[j];
        float zg = gsum[mm][128 + j] + bih_s[128 + j] + bhh_s[128 + j];
        float r  = fast_sigmoid(rg);
        float zz = fast_sigmoid(zg);
        float ng = gsum[mm][256 + j] + bih_s[256 + j] +
                   r * (ghn[mm][j] + bhh_s[256 + j]);
        float n  = fast_tanh(ng);
        float hn = (1.f - zz) * n + zz * hfs[mm][j];
        hfs[mm][j] = hn;
        hbf[mm][j] = (bf16_t)hn;
      }
    }
    if (wave == 0) __builtin_amdgcn_s_wait_tensorcnt(0);  // t+1 tile landed
    __syncthreads();
  }

  // ---- final FC + softmax ---------------------------------------------------
  if (tid < MROWS * 10) {
    int mm = tid / 10, cls = tid % 10;
    float s = b_fc[cls];
    #pragma unroll 4
    for (int j = 0; j < H; ++j) s += hfs[mm][j] * W_fc[cls * H + j];
    lg[mm][cls] = s;
  }
  __syncthreads();
  if (tid < MROWS) {
    float mx = lg[tid][0];
    #pragma unroll
    for (int c2 = 1; c2 < 10; ++c2) mx = fmaxf(mx, lg[tid][c2]);
    float e[10], sum = 0.f;
    #pragma unroll
    for (int c2 = 0; c2 < 10; ++c2) { e[c2] = __expf(lg[tid][c2] - mx); sum += e[c2]; }
    float inv = 1.f / sum;
    #pragma unroll
    for (int c2 = 0; c2 < 10; ++c2) out[(b0 + tid) * 10 + c2] = e[c2] * inv;
  }
}

extern "C" void kernel_launch(void* const* d_in, const int* in_sizes, int n_in,
                              void* d_out, int out_size, void* d_ws, size_t ws_size,
                              hipStream_t stream) {
  (void)in_sizes; (void)n_in; (void)out_size; (void)d_ws; (void)ws_size;
  const float* x    = (const float*)d_in[0];
  const float* W_ih = (const float*)d_in[1];
  const float* b_ih = (const float*)d_in[2];
  const float* W_hh = (const float*)d_in[3];
  const float* b_hh = (const float*)d_in[4];
  const float* W_fc = (const float*)d_in[5];
  const float* b_fc = (const float*)d_in[6];
  float* out = (float*)d_out;

  hipLaunchKernelGGL(gru_fused, dim3(B_TOT / MROWS), dim3(TPB), 0, stream,
                     x, W_ih, b_ih, W_hh, b_hh, W_fc, b_fc, out);
}
